// MixedMHA_82042465288949
// MI455X (gfx1250) — matrix operand order repaired and checked by reading, not compiled, not run on hardware
//
#include <hip/hip_runtime.h>
#include <hip/hip_bf16.h>

#define BDIM 1024   // hidden D
#define SDIM 1024   // seq S
#define BATCH 8
#define HEADS 16
#define HDIM 64
#define NDED 16

typedef __attribute__((ext_vector_type(16))) __bf16 v16bf;
typedef __attribute__((ext_vector_type(8)))  float  v8f;
typedef __attribute__((ext_vector_type(4)))  unsigned int v4u;
typedef __attribute__((ext_vector_type(4)))  int v4i;
typedef __attribute__((ext_vector_type(8)))  int v8i;

union AFrag { v16bf v; uint4 u[2]; };

static __device__ __forceinline__ v8f wmma_bf16(v16bf a, v16bf b, v8f c) {
  // D = A(16x32 bf16) * B(32x16 bf16) + C(16x16 f32)
  return __builtin_amdgcn_wmma_f32_16x16x32_bf16(false, a, false, b, (short)0, c, false, false);
}

static __device__ __forceinline__ v8f vzero8() {
  v8f z = {0.f,0.f,0.f,0.f,0.f,0.f,0.f,0.f};
  return z;
}

#if __has_builtin(__builtin_amdgcn_tensor_load_to_lds)
#define HAVE_TDM 1
// TDM: DMA a 2D tile (rows x row_elems of 2-byte elements, global row stride
// row_stride_elems) into LDS at byte address lds_addr. D# packed per ISA 8.3/8.4.
static __device__ __forceinline__ void tdm_load_2d_bf16(
    unsigned lds_addr, const void* gptr, unsigned rows, unsigned row_elems,
    unsigned row_stride_elems) {
  unsigned long long ga = (unsigned long long)(size_t)gptr;
  const unsigned tensor_d1 = 0x100000u;  // generous dim-1 extent (OOB guard)
  v4u g0;
  g0[0] = 1u;                                        // count=1, user descriptor
  g0[1] = lds_addr;                                  // lds_addr[31:0]
  g0[2] = (unsigned)(ga & 0xFFFFFFFFu);              // global_addr[31:0]
  g0[3] = (unsigned)((ga >> 32) & 0x1FFFFFFu) | (2u << 30);  // addr[56:32]|type=2
  v8i g1;
  g1[0] = (int)(1u << 16);                           // data_size=1 (2 bytes)
  g1[1] = (int)((row_elems & 0xFFFFu) << 16);        // tensor_dim0[15:0]
  g1[2] = (int)(((row_elems >> 16) & 0xFFFFu) | ((tensor_d1 & 0xFFFFu) << 16));
  g1[3] = (int)(((tensor_d1 >> 16) & 0xFFFFu) | ((row_elems & 0xFFFFu) << 16)); // tile_dim0
  g1[4] = (int)(rows & 0xFFFFu);                     // tile_dim1 (tile_dim2=0)
  g1[5] = (int)row_stride_elems;                     // tensor_dim0_stride[31:0]
  g1[6] = 0;
  g1[7] = 0;
  v4i gz = {0, 0, 0, 0};                             // groups 2/3: unused (2D)
#if defined(__clang_major__) && __clang_major__ >= 23
  v8i gz8 = {0, 0, 0, 0, 0, 0, 0, 0};
  __builtin_amdgcn_tensor_load_to_lds(g0, g1, gz, gz, gz8, 0);
#else
  __builtin_amdgcn_tensor_load_to_lds(g0, g1, gz, gz, 0);
#endif
}
static __device__ __forceinline__ void tdm_wait() {
#if __has_builtin(__builtin_amdgcn_s_wait_tensorcnt)
  __builtin_amdgcn_s_wait_tensorcnt(0);
#else
  asm volatile("s_wait_tensorcnt 0x0" ::: "memory");
#endif
}
#else
#define HAVE_TDM 0
#endif

// ---------------- elementwise fp32 -> bf16 ----------------
__global__ __launch_bounds__(256) void cvt_f32_bf16(const float* __restrict__ src,
                                                    __bf16* __restrict__ dst, int n) {
  int i = (blockIdx.x * 256 + threadIdx.x) * 4;
  if (i + 3 < n) {
    float4 v = *(const float4*)(src + i);
    dst[i]   = (__bf16)v.x;
    dst[i+1] = (__bf16)v.y;
    dst[i+2] = (__bf16)v.z;
    dst[i+3] = (__bf16)v.w;
  }
}

// ---------------- weight convert + transpose: Wt[o,k] = W[k,o] ----------------
__global__ __launch_bounds__(256) void cvt_transpose_w(const float* __restrict__ W,
                                                       __bf16* __restrict__ Wt) {
  int i = blockIdx.x * 256 + threadIdx.x;   // i = o*D + d
  int o = i >> 10, d = i & 1023;
  Wt[i] = (__bf16)W[(size_t)d * BDIM + o];
}

// -------- bf16 WMMA GEMM, double-buffered fragments, peeled tail --------
// C[M=8192, N=1024] = A[M,K=1024] x Bt[N,K]^T ; per-wave 16x64 strip
__global__ __launch_bounds__(128) void gemm_bf16(const __bf16* __restrict__ A,
                                                 const __bf16* __restrict__ Bt,
                                                 __bf16* __restrict__ outb,
                                                 float* __restrict__ outf) {
  const int K = BDIM;
  int wave = threadIdx.x >> 5;
  int lane = threadIdx.x & 31;
  int half = lane >> 4;
  int r    = lane & 15;
  int m0 = (blockIdx.x * 4 + wave) * 16;   // gridDim.x = 128 -> M=8192
  int n0 = blockIdx.y * 64;                // gridDim.y = 16  -> N=1024

  v8f cacc[4];
#pragma unroll
  for (int nf = 0; nf < 4; ++nf) cacc[nf] = vzero8();

  const __bf16* arow = A + (size_t)(m0 + r) * K + half * 16;
  const __bf16* brow[4];
#pragma unroll
  for (int nf = 0; nf < 4; ++nf)
    brow[nf] = Bt + (size_t)(n0 + nf * 16 + r) * K + half * 16;

  AFrag a0, a1;
  v16bf b0[4], b1[4];
  a0.u[0] = *(const uint4*)(arow + 0);
  a0.u[1] = *(const uint4*)(arow + 8);
#pragma unroll
  for (int nf = 0; nf < 4; ++nf) b0[nf] = *(const v16bf*)(brow[nf]);

  int kk = 0;
  for (; kk < K - 64; kk += 64) {
    a1.u[0] = *(const uint4*)(arow + kk + 32);
    a1.u[1] = *(const uint4*)(arow + kk + 40);
#pragma unroll
    for (int nf = 0; nf < 4; ++nf) b1[nf] = *(const v16bf*)(brow[nf] + kk + 32);
    __builtin_prefetch(arow + kk + 64, 0, 1);            // global_prefetch_b8

#pragma unroll
    for (int nf = 0; nf < 4; ++nf) cacc[nf] = wmma_bf16(a0.v, b0[nf], cacc[nf]);

    a0.u[0] = *(const uint4*)(arow + kk + 64);
    a0.u[1] = *(const uint4*)(arow + kk + 72);
#pragma unroll
    for (int nf = 0; nf < 4; ++nf) b0[nf] = *(const v16bf*)(brow[nf] + kk + 64);

#pragma unroll
    for (int nf = 0; nf < 4; ++nf) cacc[nf] = wmma_bf16(a1.v, b1[nf], cacc[nf]);
  }
  // peeled tail: kk == K-64, no further loads after the kk+32 set
  a1.u[0] = *(const uint4*)(arow + kk + 32);
  a1.u[1] = *(const uint4*)(arow + kk + 40);
#pragma unroll
  for (int nf = 0; nf < 4; ++nf) b1[nf] = *(const v16bf*)(brow[nf] + kk + 32);
#pragma unroll
  for (int nf = 0; nf < 4; ++nf) cacc[nf] = wmma_bf16(a0.v, b0[nf], cacc[nf]);
#pragma unroll
  for (int nf = 0; nf < 4; ++nf) cacc[nf] = wmma_bf16(a1.v, b1[nf], cacc[nf]);

#pragma unroll
  for (int nf = 0; nf < 4; ++nf) {
#pragma unroll
    for (int j = 0; j < 8; ++j) {
      int m = m0 + half * 8 + j;
      int n = n0 + nf * 16 + r;
      size_t idx = (size_t)m * BDIM + n;
      float v = cacc[nf][j];
      if (outf) outf[idx] = v;
      if (outb) outb[idx] = (__bf16)v;
    }
  }
}

// ---------------- dedicated-weight override for first N tokens ----------------
__global__ __launch_bounds__(256) void dedicated_kernel(
    const float* __restrict__ x,
    const float* __restrict__ Wqd, const float* __restrict__ Wkd, const float* __restrict__ Wvd,
    __bf16* __restrict__ Qb, __bf16* __restrict__ Kb, __bf16* __restrict__ Vb,
    float* __restrict__ Kout, float* __restrict__ Vout) {
  int bt = blockIdx.x;
  int b = bt / NDED, t = bt % NDED;
  int which = blockIdx.y;  // 0=q 1=k 2=v
  const float* W = (which == 0) ? Wqd : (which == 1) ? Wkd : Wvd;
  W += (size_t)t * BDIM * BDIM;
  const float* xr = x + ((size_t)b * SDIM + t) * BDIM;
  int o = threadIdx.x;
  float acc[4] = {0.f, 0.f, 0.f, 0.f};
  for (int d = 0; d < BDIM; ++d) {
    float xv = xr[d];
    const float* wr = W + (size_t)d * BDIM;
    acc[0] += xv * wr[o];
    acc[1] += xv * wr[o + 256];
    acc[2] += xv * wr[o + 512];
    acc[3] += xv * wr[o + 768];
  }
  size_t ob = ((size_t)b * SDIM + t) * BDIM;
#pragma unroll
  for (int i = 0; i < 4; ++i) {
    int oi = o + i * 256;
    if (which == 0) {
      Qb[ob + oi] = (__bf16)acc[i];
    } else if (which == 1) {
      Kb[ob + oi] = (__bf16)acc[i];
      Kout[ob + oi] = acc[i];
    } else {
      Vb[ob + oi] = (__bf16)acc[i];
      Vout[ob + oi] = acc[i];
    }
  }
}

// ---------------- V transpose: Vt[b,h,hd,s] = V[b,s,h,hd] ----------------
__global__ __launch_bounds__(256) void transpose_v(const __bf16* __restrict__ Vb,
                                                   __bf16* __restrict__ Vt) {
  size_t i = (size_t)blockIdx.x * 256 + threadIdx.x;  // over [B,S,D]
  int d = (int)(i & 1023);
  size_t bs = i >> 10;
  int s = (int)(bs & 1023);
  int b = (int)(bs >> 10);
  int h = d >> 6, hd = d & 63;
  Vt[(((size_t)(b * HEADS + h)) * HDIM + hd) * SDIM + s] = Vb[i];
}

// ---------------- causal flash attention (TDM K-tiles + bf16 WMMA) ----------------
__global__ __launch_bounds__(128) void attn_kernel(const __bf16* __restrict__ Qb,
                                                   const __bf16* __restrict__ Kb,
                                                   const __bf16* __restrict__ Vt,
                                                   __bf16* __restrict__ Ob) {
  __shared__ __bf16 pst[4][16][32];    // per-wave P re-layout staging (4 KB)
#if HAVE_TDM
  __shared__ __bf16 ktile[4][32][64];  // per-wave TDM K staging (16 KB)
#endif
  int wave = threadIdx.x >> 5;
  int lane = threadIdx.x & 31;
  int half = lane >> 4;
  int r    = lane & 15;
  int bh = blockIdx.y;
  int b = bh / HEADS, h = bh % HEADS;
  int qbase = blockIdx.x * 64 + wave * 16;

  const __bf16* Qh = Qb + ((size_t)b * SDIM) * BDIM + h * HDIM;
  const __bf16* Kh = Kb + ((size_t)b * SDIM) * BDIM + h * HDIM;
  const __bf16* Vh = Vt + ((size_t)bh * HDIM) * SDIM;

  // Q fragments (constant over the k loop)
  AFrag qf[2];
  const __bf16* qrow = Qh + (size_t)(qbase + r) * BDIM;
#pragma unroll
  for (int h2 = 0; h2 < 2; ++h2) {
    qf[h2].u[0] = *(const uint4*)(qrow + h2 * 32 + half * 16);
    qf[h2].u[1] = *(const uint4*)(qrow + h2 * 32 + half * 16 + 8);
  }

  v8f of[4];
#pragma unroll
  for (int f = 0; f < 4; ++f) of[f] = vzero8();
  float mx[8], l[8];
#pragma unroll
  for (int j = 0; j < 8; ++j) { mx[j] = -3.0e38f; l[j] = 0.f; }

#if HAVE_TDM
  unsigned klds = (unsigned)(size_t)&ktile[wave][0][0];
#endif

  int nkt = (qbase + 47) >> 5;  // causal bound, k-tiles of 32
  for (int kt = 0; kt < nkt; ++kt) {
    int kbase = kt * 32;

#if HAVE_TDM
    // TDM: async DMA of the 32x64 bf16 K-tile into this wave's LDS region.
    tdm_load_2d_bf16(klds, Kh + (size_t)kbase * BDIM, 32, HDIM, BDIM);
#endif
    // V fragments: direct global loads, in flight through softmax
    v16bf vf[4];
#pragma unroll
    for (int f = 0; f < 4; ++f)
      vf[f] = *(const v16bf*)(Vh + (size_t)(f * 16 + r) * SDIM + kbase + half * 16);

    v16bf kf[4];  // [ksub*2 + h2]
#if HAVE_TDM
    tdm_wait();   // s_wait_tensorcnt 0: K tile resident in LDS
#pragma unroll
    for (int h2 = 0; h2 < 2; ++h2) {
      kf[h2]     = *(const v16bf*)(&ktile[wave][r][h2 * 32 + half * 16]);
      kf[2 + h2] = *(const v16bf*)(&ktile[wave][16 + r][h2 * 32 + half * 16]);
    }
#else
#pragma unroll
    for (int h2 = 0; h2 < 2; ++h2) {
      kf[h2]     = *(const v16bf*)(Kh + (size_t)(kbase + r) * BDIM + h2 * 32 + half * 16);
      kf[2 + h2] = *(const v16bf*)(Kh + (size_t)(kbase + 16 + r) * BDIM + h2 * 32 + half * 16);
    }
#endif

    v8f s0 = vzero8(), s1 = vzero8();
    s0 = wmma_bf16(qf[0].v, kf[0], s0);
    s0 = wmma_bf16(qf[1].v, kf[1], s0);
    s1 = wmma_bf16(qf[0].v, kf[2], s1);
    s1 = wmma_bf16(qf[1].v, kf[3], s1);

    float p0[8], p1[8], scl[8];
#pragma unroll
    for (int j = 0; j < 8; ++j) {
      int qg = qbase + half * 8 + j;
      float v0 = s0[j] * 0.125f;
      if (kbase + r > qg) v0 = -1.0e30f;
      float v1 = s1[j] * 0.125f;
      if (kbase + 16 + r > qg) v1 = -1.0e30f;
      float t = fmaxf(v0, v1);
      t = fmaxf(t, __shfl_xor(t, 1, 32));
      t = fmaxf(t, __shfl_xor(t, 2, 32));
      t = fmaxf(t, __shfl_xor(t, 4, 32));
      t = fmaxf(t, __shfl_xor(t, 8, 32));
      float mnew = fmaxf(mx[j], t);
      float sc = __expf(mx[j] - mnew);
      float e0 = __expf(v0 - mnew);
      float e1 = __expf(v1 - mnew);
      float rs = e0 + e1;
      rs += __shfl_xor(rs, 1, 32);
      rs += __shfl_xor(rs, 2, 32);
      rs += __shfl_xor(rs, 4, 32);
      rs += __shfl_xor(rs, 8, 32);
      l[j] = l[j] * sc + rs;
      mx[j] = mnew;
      scl[j] = sc; p0[j] = e0; p1[j] = e1;
    }

    // rescale O, stage P (C-frag layout -> LDS row-major)
#pragma unroll
    for (int j = 0; j < 8; ++j) {
      of[0][j] *= scl[j]; of[1][j] *= scl[j];
      of[2][j] *= scl[j]; of[3][j] *= scl[j];
      int m = half * 8 + j;
      pst[wave][m][r]      = (__bf16)p0[j];
      pst[wave][m][16 + r] = (__bf16)p1[j];
    }
    __asm__ volatile("" ::: "memory");  // per-wave LDS RAW; DS pipe is in-order per wave

    AFrag pa;
    pa.u[0] = *(const uint4*)(&pst[wave][r][half * 16]);
    pa.u[1] = *(const uint4*)(&pst[wave][r][half * 16 + 8]);

#pragma unroll
    for (int f = 0; f < 4; ++f) of[f] = wmma_bf16(pa.v, vf[f], of[f]);
  }

  __bf16* orow = Ob + ((size_t)b * SDIM) * BDIM + h * HDIM;
#pragma unroll
  for (int j = 0; j < 8; ++j) {
    float inv = 1.0f / l[j];
    int qg = qbase + half * 8 + j;
#pragma unroll
    for (int f = 0; f < 4; ++f) {
      orow[(size_t)qg * BDIM + f * 16 + r] = (__bf16)(of[f][j] * inv);
    }
  }
}

extern "C" void kernel_launch(void* const* d_in, const int* in_sizes, int n_in,
                              void* d_out, int out_size, void* d_ws, size_t ws_size,
                              hipStream_t stream) {
  const float* x   = (const float*)d_in[0];
  const float* Wq  = (const float*)d_in[1];
  const float* Wk  = (const float*)d_in[2];
  const float* Wv  = (const float*)d_in[3];
  const float* Wqd = (const float*)d_in[4];
  const float* Wkd = (const float*)d_in[5];
  const float* Wvd = (const float*)d_in[6];
  const float* Wo  = (const float*)d_in[7];

  float* out  = (float*)d_out;                         // [B,S,D]
  float* kout = out + (size_t)BATCH * SDIM * BDIM;     // [B,S,H,HD] flat
  float* vout = kout + (size_t)BATCH * SDIM * BDIM;

  const size_t XE = (size_t)BATCH * SDIM * BDIM;  // 8388608
  const size_t WE = (size_t)BDIM * BDIM;          // 1048576
  __bf16* ws  = (__bf16*)d_ws;
  __bf16* Xb  = ws;
  __bf16* Wqt = Xb + XE;
  __bf16* Wkt = Wqt + WE;
  __bf16* Wvt = Wkt + WE;
  __bf16* Wot = Wvt + WE;
  __bf16* Qb  = Wot + WE;
  __bf16* Kb  = Qb + XE;
  __bf16* Vb  = Kb + XE;
  __bf16* Vt  = Vb + XE;
  __bf16* Ob  = Vt + XE;   // total ~104 MB of workspace

  // 1) precision conversion (+ weight transpose for WMMA B-fragment locality)
  cvt_f32_bf16<<<(unsigned)(XE / 4 / 256), 256, 0, stream>>>(x, Xb, (int)XE);
  cvt_transpose_w<<<(unsigned)(WE / 256), 256, 0, stream>>>(Wq, Wqt);
  cvt_transpose_w<<<(unsigned)(WE / 256), 256, 0, stream>>>(Wk, Wkt);
  cvt_transpose_w<<<(unsigned)(WE / 256), 256, 0, stream>>>(Wv, Wvt);
  cvt_transpose_w<<<(unsigned)(WE / 256), 256, 0, stream>>>(Wo, Wot);

  // 2) shared projections (K/V dual-write fp32 outputs)
  gemm_bf16<<<dim3(128, 16), 128, 0, stream>>>(Xb, Wqt, Qb, nullptr);
  gemm_bf16<<<dim3(128, 16), 128, 0, stream>>>(Xb, Wkt, Kb, kout);
  gemm_bf16<<<dim3(128, 16), 128, 0, stream>>>(Xb, Wvt, Vb, vout);

  // 3) dedicated-weight override for the first N tokens
  dedicated_kernel<<<dim3(BATCH * NDED, 3), 256, 0, stream>>>(
      x, Wqd, Wkd, Wvd, Qb, Kb, Vb, kout, vout);

  // 4) V transpose for P*V B-fragments
  transpose_v<<<(unsigned)(XE / 256), 256, 0, stream>>>(Vb, Vt);

  // 5) causal flash attention
  attn_kernel<<<dim3(SDIM / 64, BATCH * HEADS), 128, 0, stream>>>(Qb, Kb, Vt, Ob);

  // 6) output projection
  gemm_bf16<<<dim3(128, 16), 128, 0, stream>>>(Ob, Wot, nullptr, out);
}
